// Critic_50276887167257
// MI455X (gfx1250) — matrix operand order repaired and checked by reading, not compiled, not run on hardware
//
#include <hip/hip_runtime.h>
#include <hip/hip_bf16.h>

typedef __attribute__((ext_vector_type(16))) __bf16 v16bf;
typedef __attribute__((ext_vector_type(8)))  __bf16 v8bf;
typedef __attribute__((ext_vector_type(8)))  float  v8f;

#define HCH 128   // hidden channels

// ---------------- small utility kernels ----------------
__global__ void k_zero(float* p, int n) {
    int i = blockIdx.x * blockDim.x + threadIdx.x;
    if (i < n) p[i] = 0.f;
}

// Pack W[K,128] f32 -> bf16 in WMMA B-fragment order:
// Bf[((kt*8 + ct)*32 + lane)*16 + i] = (bf16)W[(kt*32 + (i<8 ? (lane>>4)*8 + i : 16 + (lane>>4)*8 + i - 8))*128 + ct*16 + (lane&15)]
// so each lane's 16 fragment values are one contiguous 32-byte chunk.
__global__ void k_pack_b(const float* __restrict__ W, __bf16* __restrict__ Bf, int K) {
    int t = blockIdx.x * blockDim.x + threadIdx.x;
    int total = (K / 32) * 8 * 32;
    if (t >= total) return;
    int lane = t & 31;
    int ct   = (t >> 5) & 7;
    int kt   = t >> 8;
    int half = lane >> 4, n = lane & 15;
    v16bf v;
#pragma unroll
    for (int i = 0; i < 16; ++i) {
        int k = kt * 32 + (i < 8 ? half * 8 + i : 16 + half * 8 + (i - 8));
        v[i] = (__bf16)W[(size_t)k * HCH + ct * 16 + n];
    }
    ((v16bf*)Bf)[t] = v;
}

__global__ void k_deg_init(float* deg, int n) {
    int i = blockIdx.x * blockDim.x + threadIdx.x;
    if (i < n) deg[i] = 1.f;            // self loop
}

__global__ void k_deg_count(const int* __restrict__ ei, float* deg, int E) {
    int e = blockIdx.x * blockDim.x + threadIdx.x;
    if (e < E) atomicAdd(&deg[ei[E + e]], 1.f);   // dst = second row of edge_index
}

__global__ void k_rsqrt_inplace(float* d, int n) {
    int i = blockIdx.x * blockDim.x + threadIdx.x;
    if (i < n) d[i] = rsqrtf(d[i]);
}

// h0acc[i] = xw[i] * dinv[node]^2   (self-loop term; also initializes the accumulator)
__global__ void k_selfloop(const float* __restrict__ xw, const float* __restrict__ dinv,
                           float* __restrict__ acc, int total) {
    int i = blockIdx.x * blockDim.x + threadIdx.x;
    if (i < total) {
        int node = i >> 7;
        float di = dinv[node];
        acc[i] = xw[i] * di * di;
    }
}

// one wave per edge, each lane scatters 4 channels (float4 gather stays in L2: xw is 102MB < 192MB)
__global__ void k_edge_scatter(const int* __restrict__ ei, const float* __restrict__ dinv,
                               const float* __restrict__ xw, float* __restrict__ acc, int E) {
    int t = blockIdx.x * blockDim.x + threadIdx.x;
    int e = t >> 5;
    if (e >= E) return;
    int lane = t & 31;
    int src = ei[e];
    int dst = ei[E + e];
    float nrm = dinv[src] * dinv[dst];
    float4 v = ((const float4*)(xw + (size_t)src * HCH))[lane];
    float* a = acc + (size_t)dst * HCH + lane * 4;
    atomicAdd(a + 0, v.x * nrm);
    atomicAdd(a + 1, v.y * nrm);
    atomicAdd(a + 2, v.z * nrm);
    atomicAdd(a + 3, v.w * nrm);
}

__global__ void k_bias_relu(float* h, const float* __restrict__ b, int total) {
    int i = blockIdx.x * blockDim.x + threadIdx.x;
    if (i < total) {
        int c = i & 127;
        h[i] = fmaxf(h[i] + b[c], 0.f);
    }
}

// per-channel sum / sum-of-squares (for BN batch stats)
__global__ void k_stats(const float* __restrict__ h, int nrows, float* sum, float* sq) {
    int tid = blockIdx.x * blockDim.x + threadIdx.x;
    int c = tid & 127;
    int r0 = tid >> 7;
    int stride = (gridDim.x * blockDim.x) >> 7;
    float s = 0.f, q = 0.f;
    for (int r = r0; r < nrows; r += stride) {
        float v = h[(size_t)r * HCH + c];
        s += v; q += v * v;
    }
    atomicAdd(&sum[c], s);
    atomicAdd(&sq[c], q);
}

__global__ void k_bn_finalize(const float* sum, const float* sq,
                              const float* __restrict__ g, const float* __restrict__ be,
                              float* scale, float* shift, float invN) {
    int c = threadIdx.x;
    if (c < 128) {
        float mu  = sum[c] * invN;
        float var = sq[c] * invN - mu * mu;
        float s   = g[c] * rsqrtf(var + 1e-5f);
        scale[c]  = s;
        shift[c]  = be[c] - mu * s;
    }
}

// ---------------- WMMA GEMM: Out[N,128] = relu?(BN?(A[N,K]) @ W[K,128] + bias) ----------------
// block = 128 threads = 4 waves; block does a 16-row stripe, each wave two 16x16 N-tiles.
// ASYNC=1: A tile staged f32 via global_load_async_to_lds_b128 (ASYNCcnt), cvt in frag build.
// ASYNC=0: A tile staged bf16 with BN affine fused at staging time.
template <int K, bool USE_BN, bool BIAS_RELU, bool ASYNC>
__global__ __launch_bounds__(128)
void k_gemm(const float* __restrict__ A,
            const float* __restrict__ scale, const float* __restrict__ shift,
            const __bf16* __restrict__ Wb,     // packed fragment-order weights
            const float* __restrict__ bias,
            float* __restrict__ Out) {
    constexpr int SF = K + 4;                  // f32 LDS row stride (pads to conflict-free b128)
    constexpr int SB = K + 8;                  // bf16 LDS row stride
    constexpr int SMEM = ASYNC ? 16 * SF * 4 : 16 * SB * 2;
    __shared__ __align__(16) unsigned char smem[SMEM];
    float*  AsF = (float*)smem;
    __bf16* AsB = (__bf16*)smem;

    const int tid = threadIdx.x;
    const int rowBase = blockIdx.x * 16;

    if constexpr (ASYNC) {
        // async copy 16 rows x K f32, 16B chunks per lane; LDS offset = low 32 bits of AS(3) addr
        constexpr int CPR = K / 4;             // float4 chunks per row
        for (int i = tid; i < 16 * CPR; i += 128) {
            int r = i / CPR, c4 = i - r * CPR;
            unsigned lds_off = (unsigned)(unsigned long long)(void*)&AsF[r * SF + c4 * 4];
            const float* g = A + (size_t)(rowBase + r) * K + c4 * 4;
            asm volatile("global_load_async_to_lds_b128 %0, %1, off"
                         :: "v"(lds_off), "v"(g) : "memory");
        }
        asm volatile("s_wait_asynccnt 0x0" ::: "memory");
        __syncthreads();
    } else {
        for (int i = tid; i < 16 * K; i += 128) {      // coalesced, BN + bf16-cvt fused
            int r = i / K, c = i - r * K;
            float v = A[(size_t)(rowBase + r) * K + c];
            if constexpr (USE_BN) v = v * scale[c] + shift[c];
            AsB[r * SB + c] = (__bf16)v;
        }
        __syncthreads();
    }

    const int lane = tid & 31;
    const int wave = tid >> 5;
    const int half = lane >> 4;     // 0: lanes 0-15, 1: lanes 16-31
    const int mn   = lane & 15;     // A row / B,C col within 16x16 tile
    const int n0   = wave * 32;
    const v16bf* Bf = (const v16bf*)Wb;

    v8f c0 = {}; v8f c1 = {};
#pragma unroll
    for (int kk = 0; kk < K; kk += 32) {
        // A 16x32 bf16 layout: per lane, two runs of 8 consecutive K values:
        // [kk+half*8 .. +7] and [kk+16+half*8 .. +7]
        v16bf a;
        if constexpr (ASYNC) {
            const float* rp = AsF + mn * SF + kk + half * 8;
            float4 f0 = *(const float4*)(rp);
            float4 f1 = *(const float4*)(rp + 4);
            float4 f2 = *(const float4*)(rp + 16);
            float4 f3 = *(const float4*)(rp + 20);
            a[0]=(__bf16)f0.x; a[1]=(__bf16)f0.y; a[2]=(__bf16)f0.z; a[3]=(__bf16)f0.w;
            a[4]=(__bf16)f1.x; a[5]=(__bf16)f1.y; a[6]=(__bf16)f1.z; a[7]=(__bf16)f1.w;
            a[8]=(__bf16)f2.x; a[9]=(__bf16)f2.y; a[10]=(__bf16)f2.z; a[11]=(__bf16)f2.w;
            a[12]=(__bf16)f3.x; a[13]=(__bf16)f3.y; a[14]=(__bf16)f3.z; a[15]=(__bf16)f3.w;
        } else {
            const __bf16* rp = AsB + mn * SB + kk + half * 8;
            v8bf lo = *(const v8bf*)(rp);
            v8bf hi = *(const v8bf*)(rp + 16);
#pragma unroll
            for (int i = 0; i < 8; ++i) { a[i] = lo[i]; a[8 + i] = hi[i]; }
        }
        // packed B fragments: one 32-byte contiguous load per tile
        int kt = kk >> 5;
        v16bf b0 = Bf[(size_t)((kt * 8 + 2 * wave)     * 32) + lane];
        v16bf b1 = Bf[(size_t)((kt * 8 + 2 * wave + 1) * 32) + lane];
        c0 = __builtin_amdgcn_wmma_f32_16x16x32_bf16(false, a, false, b0, (short)0, c0, false, false);
        c1 = __builtin_amdgcn_wmma_f32_16x16x32_bf16(false, a, false, b1, (short)0, c1, false, false);
    }

#pragma unroll
    for (int v = 0; v < 8; ++v) {                      // C/D: VGPR v -> row v + half*8
        int r = rowBase + v + half * 8;
        int col0 = n0 + mn, col1 = n0 + 16 + mn;
        float x0 = c0[v], x1 = c1[v];
        if constexpr (BIAS_RELU) {
            x0 = fmaxf(x0 + bias[col0], 0.f);
            x1 = fmaxf(x1 + bias[col1], 0.f);
        }
        Out[(size_t)r * HCH + col0] = x0;
        Out[(size_t)r * HCH + col1] = x1;
    }
}

// fold W3 with BN1/BN2 affines: a1[c]=scale1[c]*W3[c], a2[c]=scale2[c]*W3[128+c],
// constS = sum(shift1*W3[:128]) + sum(shift2*W3[128:])
__global__ void k_combine(const float* scale1, const float* shift1,
                          const float* scale2, const float* shift2,
                          const float* __restrict__ W3,
                          float* a1, float* a2, float* constS) {
    __shared__ float red[256];
    int t = threadIdx.x;
    float p;
    if (t < 128) { float w = W3[t];               a1[t] = scale1[t] * w; p = shift1[t] * w; }
    else         { int c = t - 128; float w = W3[128 + c]; a2[c] = scale2[c] * w; p = shift2[c] * w; }
    red[t] = p; __syncthreads();
    for (int s = 128; s > 0; s >>= 1) { if (t < s) red[t] += red[t + s]; __syncthreads(); }
    if (t == 0) constS[0] = red[0];
}

// one wave per node: s_n = h1[n]·a1 + h2[n]·a2 + constS ; atomic segment-sum by batch id
__global__ __launch_bounds__(256)
void k_pool(const float* __restrict__ h1, const float* __restrict__ h2,
            const float* __restrict__ a1, const float* __restrict__ a2,
            const float* __restrict__ constS, const int* __restrict__ batch,
            float* bsum, float* bcnt, int nrows) {
    int wave = threadIdx.x >> 5, lane = threadIdx.x & 31;
    int node = blockIdx.x * 8 + wave;
    if (node >= nrows) return;
    const float* r1 = h1 + (size_t)node * HCH;
    const float* r2 = h2 + (size_t)node * HCH;
    float s = 0.f;
#pragma unroll
    for (int c = lane; c < 128; c += 32) s += r1[c] * a1[c] + r2[c] * a2[c];
    for (int off = 16; off > 0; off >>= 1) s += __shfl_down(s, off, 32);
    if (lane == 0) {
        int bi = batch[node];
        atomicAdd(&bsum[bi], s + constS[0]);
        atomicAdd(&bcnt[bi], 1.f);
    }
}

__global__ void k_out(const float* bsum, const float* bcnt, const float* __restrict__ b3,
                      float* out, int B) {
    int b = blockIdx.x * blockDim.x + threadIdx.x;
    if (b < B) out[b] = bsum[b] / fmaxf(bcnt[b], 1.f) + b3[0];
}

// ---------------- launch ----------------
extern "C" void kernel_launch(void* const* d_in, const int* in_sizes, int n_in,
                              void* d_out, int out_size, void* d_ws, size_t ws_size,
                              hipStream_t stream) {
    (void)in_sizes; (void)n_in; (void)out_size; (void)ws_size;
    const int N = 200000, E = 1600000, B = 1000, SD = 64, AD = 32, H = HCH;

    const float* x      = (const float*)d_in[0];
    const float* action = (const float*)d_in[1];
    const float* W_gcn  = (const float*)d_in[2];
    const float* b_gcn  = (const float*)d_in[3];
    const float* g0     = (const float*)d_in[4];
    const float* be0    = (const float*)d_in[5];
    const float* W1     = (const float*)d_in[6];
    const float* b1     = (const float*)d_in[7];
    const float* g1     = (const float*)d_in[8];
    const float* be1    = (const float*)d_in[9];
    const float* W2     = (const float*)d_in[10];
    const float* b2     = (const float*)d_in[11];
    const float* g2     = (const float*)d_in[12];
    const float* be2    = (const float*)d_in[13];
    const float* W3     = (const float*)d_in[14];
    const float* b3     = (const float*)d_in[15];
    const int*   ei     = (const int*)d_in[16];
    const int*   batch  = (const int*)d_in[17];
    float* out = (float*)d_out;

    char* ws = (char*)d_ws;
    size_t off = 0;
    auto alloc = [&](size_t bytes) -> char* {
        char* p = ws + off;
        off = (off + bytes + 255) & ~(size_t)255;
        return p;
    };
    float*  buf1   = (float*)alloc((size_t)N * H * 4);   // xw, then h1
    float*  buf2   = (float*)alloc((size_t)N * H * 4);   // h0acc -> h0, then h2
    float*  dinv   = (float*)alloc((size_t)N * 4);
    __bf16* wgcn_b = (__bf16*)alloc((size_t)SD * H * 2); // packed fragment order
    __bf16* w1_b   = (__bf16*)alloc((size_t)H * H * 2);
    __bf16* w2_b   = (__bf16*)alloc((size_t)AD * H * 2);
    float*  zr     = (float*)alloc((size_t)(768 + 2 * B) * 4);  // zeroed each call
    float* sum0 = zr;       float* sq0 = zr + 128;
    float* sum1 = zr + 256; float* sq1 = zr + 384;
    float* sum2 = zr + 512; float* sq2 = zr + 640;
    float* bsum = zr + 768; float* bcnt = zr + 768 + B;
    float* scale0 = (float*)alloc(128 * 4); float* shift0 = (float*)alloc(128 * 4);
    float* scale1 = (float*)alloc(128 * 4); float* shift1 = (float*)alloc(128 * 4);
    float* scale2 = (float*)alloc(128 * 4); float* shift2 = (float*)alloc(128 * 4);
    float* a1 = (float*)alloc(128 * 4);
    float* a2 = (float*)alloc(128 * 4);
    float* cS = (float*)alloc(4);

    const float invN = 1.f / (float)N;
    const int total = N * H;

    k_zero<<<(768 + 2 * B + 255) / 256, 256, 0, stream>>>(zr, 768 + 2 * B);
    k_pack_b<<<((SD / 32) * 256 + 255) / 256, 256, 0, stream>>>(W_gcn, wgcn_b, SD);
    k_pack_b<<<((H  / 32) * 256 + 255) / 256, 256, 0, stream>>>(W1, w1_b, H);
    k_pack_b<<<((AD / 32) * 256 + 255) / 256, 256, 0, stream>>>(W2, w2_b, AD);

    // degrees -> dinv
    k_deg_init<<<(N + 255) / 256, 256, 0, stream>>>(dinv, N);
    k_deg_count<<<(E + 255) / 256, 256, 0, stream>>>(ei, dinv, E);
    k_rsqrt_inplace<<<(N + 255) / 256, 256, 0, stream>>>(dinv, N);

    // xw = x @ W_gcn  (no bias yet) — async-LDS staged
    k_gemm<64, false, false, true><<<N / 16, 128, 0, stream>>>(
        x, nullptr, nullptr, wgcn_b, nullptr, buf1);
    // GCN aggregate: self loops init accumulator, then edge scatter
    k_selfloop<<<total / 256, 256, 0, stream>>>(buf1, dinv, buf2, total);
    k_edge_scatter<<<(E * 32) / 256, 256, 0, stream>>>(ei, dinv, buf1, buf2, E);
    k_bias_relu<<<total / 256, 256, 0, stream>>>(buf2, b_gcn, total);

    // BN0 stats + finalize
    k_stats<<<512, 256, 0, stream>>>(buf2, N, sum0, sq0);
    k_bn_finalize<<<1, 128, 0, stream>>>(sum0, sq0, g0, be0, scale0, shift0, invN);

    // h1 = relu(BN0(h0) @ W1 + b1)   (BN fused into bf16 staging)
    k_gemm<128, true, true, false><<<N / 16, 128, 0, stream>>>(
        buf2, scale0, shift0, w1_b, b1, buf1);
    k_stats<<<512, 256, 0, stream>>>(buf1, N, sum1, sq1);

    // h2 = relu(action @ W2 + b2) — async-LDS staged
    k_gemm<32, false, true, true><<<N / 16, 128, 0, stream>>>(
        action, nullptr, nullptr, w2_b, b2, buf2);
    k_stats<<<512, 256, 0, stream>>>(buf2, N, sum2, sq2);

    k_bn_finalize<<<1, 128, 0, stream>>>(sum1, sq1, g1, be1, scale1, shift1, invN);
    k_bn_finalize<<<1, 128, 0, stream>>>(sum2, sq2, g2, be2, scale2, shift2, invN);

    // fold W3 through BN1/BN2, then segment-mean of per-node scalars
    k_combine<<<1, 256, 0, stream>>>(scale1, shift1, scale2, shift2, W3, a1, a2, cS);
    k_pool<<<(N + 7) / 8, 256, 0, stream>>>(buf1, buf2, a1, a2, cS, batch, bsum, bcnt, N);
    k_out<<<(B + 255) / 256, 256, 0, stream>>>(bsum, bcnt, b3, out, B);
}
